// FasterRCNNTrainer_42494406427385
// MI455X (gfx1250) — matrix operand order repaired and checked by reading, not compiled, not run on hardware
//
#include <hip/hip_runtime.h>
#include <hip/hip_bf16.h>
#include <math.h>

// Problem constants (match the reference)
#define NB   4
#define NT   40
#define NR   2000
#define NA   36864
#define NH   64
#define NW   64
#define NC   256
#define NPP  7
#define ND   (NPP*NPP*NC)   // 12544
#define NDH  1024
#define NCLS 21
#define NPOS 64
#define NNEG 192
#define NROI 256            // NPOS + NNEG
#define NRC  32             // padded combined [reg(4) | cls(21) | pad(7)] columns

typedef __attribute__((ext_vector_type(16))) _Float16 v16h;
typedef __attribute__((ext_vector_type(8)))  _Float16 v8h;
typedef __attribute__((ext_vector_type(8)))  float    v8f;

__device__ __forceinline__ float iou_box(float t1,float l1,float b1,float r1,
                                         float t2,float l2,float b2,float r2)
{
    float t = fmaxf(t1, t2), l = fmaxf(l1, l2);
    float b = fminf(b1, b2), r = fminf(r1, r2);
    float ih = b - t; ih = ih > 0.f ? ih : 0.f;
    float iw = r - l; iw = iw > 0.f ? iw : 0.f;
    float inter = ih * iw;
    float a1 = (b1 - t1) * (r1 - l1);
    float a2 = (b2 - t2) * (r2 - l2);
    return inter / (a1 + a2 - inter);
}

// ---- (1) IoU of bboxes vs anchors: max / argmax over T (axis=1) -------------
__global__ void k_rpn_iou(const float* __restrict__ bboxes, const float* __restrict__ anchors,
                          float* __restrict__ max_iou, int* __restrict__ tidx)
{
    int g = blockIdx.x * blockDim.x + threadIdx.x;   // g = b*NA + a
    if (g >= NB * NA) return;
    int b = g / NA, a = g - b * NA;
    const float* an = anchors + (size_t)a * 4;
    float at = an[0], al = an[1], ab = an[2], ar = an[3];
    const float* bb = bboxes + (size_t)b * NT * 4;
    float best = -1.f; int bi = 0;
    for (int t = 0; t < NT; ++t) {
        float v = iou_box(bb[t*4+0], bb[t*4+1], bb[t*4+2], bb[t*4+3], at, al, ab, ar);
        if (v > best) { best = v; bi = t; }          // first-max tie break like argmax
    }
    max_iou[g] = best; tidx[g] = bi;
}

// ---- (2) IoU of nms boxes vs bboxes: max / argmax over T, plus neg score ----
__global__ void k_nms_iou(const float* __restrict__ nms, const float* __restrict__ bboxes,
                          float* __restrict__ iou_m, int* __restrict__ ridx,
                          float* __restrict__ negsc)
{
    int g = blockIdx.x * blockDim.x + threadIdx.x;   // g = b*NR + r
    if (g >= NB * NR) return;
    int b = g / NR;
    const float* bx = nms + (size_t)g * 4;
    float nt = bx[0], nl = bx[1], nb = bx[2], nr = bx[3];
    const float* bb = bboxes + (size_t)b * NT * 4;
    float best = -1.f; int bi = 0;
    for (int t = 0; t < NT; ++t) {
        float v = iou_box(nt, nl, nb, nr, bb[t*4+0], bb[t*4+1], bb[t*4+2], bb[t*4+3]);
        if (v > best) { best = v; bi = t; }
    }
    iou_m[g] = best; ridx[g] = bi;
    negsc[g] = (best <= 0.5f) ? best : -__builtin_inff();   // jnp.where(iou<=.5, iou, -inf)
}

// ---- first-k ascending selection (== jnp.nonzero(mask, size=k), 0-padded) ---
// Single block, 1024 threads; packed (pos<<16 | neg) block scans.
__global__ void k_select_first(const float* __restrict__ vals, int n, float thr,
                               int* __restrict__ out_pos, int kpos,
                               int* __restrict__ out_neg, int kneg)
{
    __shared__ int scan[1024];
    __shared__ int s_base;
    int tid = threadIdx.x;
    if (tid < kpos) out_pos[tid] = 0;   // nonzero(size=k) pads with 0
    if (tid < kneg) out_neg[tid] = 0;
    if (tid == 0) s_base = 0;
    __syncthreads();
    int nchunk = (n + 1023) / 1024;
    for (int chunk = 0; chunk < nchunk; ++chunk) {
        int i = chunk * 1024 + tid;
        int p = 0, q = 0;
        if (i < n) { p = (vals[i] > thr) ? 1 : 0; q = 1 - p; }
        int f = (p << 16) | q;
        scan[tid] = f;
        __syncthreads();
        for (int off = 1; off < 1024; off <<= 1) {
            int add = (tid >= off) ? scan[tid - off] : 0;
            __syncthreads();
            scan[tid] += add;
            __syncthreads();
        }
        int incl = scan[tid];
        int excl = incl - f;
        int basep = s_base >> 16, baseq = s_base & 0xffff;
        if (p) { int idx = basep + (excl >> 16);    if (idx < kpos) out_pos[idx] = i; }
        if (q) { int idx = baseq + (excl & 0xffff); if (idx < kneg) out_neg[idx] = i; }
        __syncthreads();
        if (tid == 1023) {
            int tp = basep + (scan[1023] >> 16);    if (tp > 60000) tp = 60000;
            int tq = baseq + (scan[1023] & 0xffff); if (tq > 60000) tq = 60000;
            s_base = (tp << 16) | tq;
        }
        __syncthreads();
        if ((s_base >> 16) >= kpos && (s_base & 0xffff) >= kneg) break;  // uniform
    }
}

// ---- top-k descending (ties -> lower index), iterative; mutates score ------
__global__ void k_topk(float* __restrict__ score, int n, int* __restrict__ out, int k)
{
    __shared__ float sv[1024];
    __shared__ int   si[1024];
    int tid = threadIdx.x;
    for (int it = 0; it < k; ++it) {
        float bv = -__builtin_inff(); int bi = 0x7fffffff;
        for (int i = tid; i < n; i += 1024) {
            float v = score[i];
            if (v > bv || (v == bv && i < bi)) { bv = v; bi = i; }
        }
        sv[tid] = bv; si[tid] = bi;
        __syncthreads();
        for (int off = 512; off > 0; off >>= 1) {
            if (tid < off) {
                float ov = sv[tid + off]; int oi = si[tid + off];
                if (ov > sv[tid] || (ov == sv[tid] && oi < si[tid])) { sv[tid] = ov; si[tid] = oi; }
            }
            __syncthreads();
        }
        if (tid == 0) { out[it] = si[0]; score[si[0]] = -__builtin_inff(); }
        __syncthreads();
    }
}

// ---- RPN losses (BCE over 380 logits; smooth-L1 over 128x4) ----------------
__global__ void k_rpn_losses(const int* __restrict__ pos, const int* __restrict__ neg,
                             const float* __restrict__ rpn_cls, const float* __restrict__ rpn_reg,
                             const float* __restrict__ bboxes, const int* __restrict__ tidx,
                             const float* __restrict__ anchors, float* __restrict__ out)
{
    __shared__ float s1[512];
    __shared__ float s2[512];
    int i = threadIdx.x;
    float bce = 0.f;
    if (i < 380) {
        float x, y;
        if (i < 128) { x = rpn_cls[pos[i]];       y = 1.f; }
        else         { x = rpn_cls[neg[i - 128]]; y = 0.f; }
        bce = fmaxf(x, 0.f) - x * y + log1pf(expf(-fabsf(x)));
    }
    int p = pos[i >> 2];
    int c = i & 3;
    int b = p / NA, a = p - b * NA;
    float pred = rpn_reg[(size_t)p * 4 + c];
    float t = bboxes[((size_t)b * NT + tidx[p]) * 4 + c] - anchors[(size_t)a * 4 + c];
    float d = fabsf(pred - t);
    float sl = (d < 1.f) ? 0.5f * d * d : (d - 0.5f);
    s1[i] = bce; s2[i] = sl;
    __syncthreads();
    for (int off = 256; off > 0; off >>= 1) {
        if (i < off) { s1[i] += s1[i + off]; s2[i] += s2[i + off]; }
        __syncthreads();
    }
    if (i == 0) { out[0] = s1[0] / 380.f; out[1] = s2[0] / 512.f / 4.f; }
}

// ---- ROI assembly: rois, class targets, reg targets ------------------------
__global__ void k_build_rois(const int* __restrict__ posr, const int* __restrict__ negr,
                             const float* __restrict__ nms, const int* __restrict__ classes,
                             const int* __restrict__ ridx, const float* __restrict__ bboxes,
                             float* __restrict__ rois, int* __restrict__ cls_s,
                             float* __restrict__ tgt)
{
    int i = threadIdx.x;                  // 0..255
    int s = (i < NPOS) ? posr[i] : negr[i - NPOS];
    int b = s / NR;
    float x0 = nms[(size_t)s*4+0], x1 = nms[(size_t)s*4+1];
    float x2 = nms[(size_t)s*4+2], x3 = nms[(size_t)s*4+3];
    rois[i*5+0] = (float)b;
    rois[i*5+1] = x0; rois[i*5+2] = x1; rois[i*5+3] = x2; rois[i*5+4] = x3;
    int rid = ridx[s];
    cls_s[i] = (i < NPOS) ? classes[b * NT + rid] : 0;
    if (i < NPOS) {
        float r0 = floorf(x0 * 16.f) / 16.f;
        float r1 = floorf(x1 * 16.f) / 16.f;
        float r2 = ceilf (x2 * 16.f) / 16.f;
        float r3 = ceilf (x3 * 16.f) / 16.f;
        const float* mb = bboxes + ((size_t)b * NT + rid) * 4;
        tgt[i*4+0] = mb[0] - r0;
        tgt[i*4+1] = mb[1] - r1;
        tgt[i*4+2] = mb[2] - r2;
        tgt[i*4+3] = mb[3] - r3;
    }
}

// ---- ROI bilinear pool, emits f16 feature rows (GEMM1 A panel) -------------
__global__ void k_roi_pool(const float* __restrict__ fm, const float* __restrict__ rois,
                           _Float16* __restrict__ feat16)
{
    int j   = blockIdx.x;     // roi
    int bin = blockIdx.y;     // gy*7+gx
    int c   = threadIdx.x;    // channel
    int gy = bin / NPP, gx = bin - gy * NPP;
    int n = (int)rois[j*5+0];
    float t = rois[j*5+1], l = rois[j*5+2], bb = rois[j*5+3], r = rois[j*5+4];
    float y = t + (((float)gy + 0.5f) * (1.0f / NPP)) * (bb - t);
    float x = l + (((float)gx + 0.5f) * (1.0f / NPP)) * (r - l);
    y = fminf(fmaxf(y, 0.f), (float)(NH - 1));
    x = fminf(fmaxf(x, 0.f), (float)(NW - 1));
    int y0 = (int)floorf(y), x0 = (int)floorf(x);
    int y1 = min(y0 + 1, NH - 1), x1 = min(x0 + 1, NW - 1);
    float wy = y - (float)y0, wx = x - (float)x0;
    const float* base = fm + ((size_t)n * NC + c) * NH * NW;
    float v00 = base[y0*NW + x0], v01 = base[y0*NW + x1];
    float v10 = base[y1*NW + x0], v11 = base[y1*NW + x1];
    float v = v00*(1.f-wy)*(1.f-wx) + v01*(1.f-wy)*wx + v10*wy*(1.f-wx) + v11*wy*wx;
    feat16[(size_t)j * ND + bin * NC + c] = (_Float16)v;
}

// ---- W1 (K x N fp32) -> W1t (N x K f16), tiled LDS transpose ----------------
__global__ void k_w1_to_f16t(const float* __restrict__ W, _Float16* __restrict__ Wt)
{
    __shared__ float tile[32][33];
    int k0 = blockIdx.x * 32, n0 = blockIdx.y * 32;
    int tx = threadIdx.x & 31, ty = threadIdx.x >> 5;      // 32x8 block
    for (int yy = ty; yy < 32; yy += 8)
        tile[yy][tx] = W[(size_t)(k0 + yy) * NDH + (n0 + tx)];
    __syncthreads();
    for (int yy = ty; yy < 32; yy += 8)
        Wt[(size_t)(n0 + yy) * ND + (k0 + tx)] = (_Float16)tile[tx][yy];
}

// ---- pack [Wr | Wc | 0] -> Wrct (32 x 1024, N-major f16) -------------------
__global__ void k_pack_wrc(const float* __restrict__ Wr, const float* __restrict__ Wc,
                           _Float16* __restrict__ Wt)
{
    int n = blockIdx.x;       // 0..31
    int k = threadIdx.x;      // 0..1023
    float v = 0.f;
    if (n < 4)            v = Wr[(size_t)k * 4 + n];
    else if (n < 4+NCLS)  v = Wc[(size_t)k * NCLS + (n - 4)];
    Wt[(size_t)n * NDH + k] = (_Float16)v;
}

// ---- GEMM1: hmid = relu(feat @ W1 + b1), f16 in / f32 acc / f16 out --------
// grid (16 m-tiles, 8), 8 waves/block, one 16x16 tile per wave.
__global__ __launch_bounds__(256) void k_gemm1(const _Float16* __restrict__ A,
                                               const _Float16* __restrict__ Bt,
                                               const float* __restrict__ bias,
                                               _Float16* __restrict__ H)
{
    const int K = ND;
    int wave = threadIdx.x >> 5, lane = threadIdx.x & 31;
    int laneHi = lane >> 4, l16 = lane & 15;
    int mt = blockIdx.x;
    int nt = blockIdx.y * 8 + wave;
    const _Float16* ap = A  + (size_t)(mt * 16 + l16) * K + laneHi * 16;
    const _Float16* bp = Bt + (size_t)(nt * 16 + l16) * K + laneHi * 16;
    v8f acc = {};
    for (int k0 = 0; k0 < K; k0 += 32) {
        v8h alo = *(const v8h*)(ap + k0);
        v8h ahi = *(const v8h*)(ap + k0 + 8);
        v16h bfr = *(const v16h*)(bp + k0);
        v16h a;
        #pragma unroll
        for (int t = 0; t < 8; ++t) { a[t] = alo[t]; a[t + 8] = ahi[t]; }
        __builtin_prefetch(bp + k0 + 512, 0, 0);   // global_prefetch_b8 down the K stream
        acc = __builtin_amdgcn_wmma_f32_16x16x32_f16(false, a, false, bfr,
                                                     (short)0, acc, false, false);
    }
    #pragma unroll
    for (int i = 0; i < 8; ++i) {
        int m = mt * 16 + i + laneHi * 8;         // C/D VGPR layout (ISA 7.12.2)
        int n = nt * 16 + l16;
        float v = acc[i] + bias[n];
        v = v > 0.f ? v : 0.f;
        H[(size_t)m * NDH + n] = (_Float16)v;
    }
}

// ---- GEMM2: rrc = hmid @ [Wr|Wc|0] + [br|bc|0], f32 out --------------------
// 32 tiles total, 8 waves/block -> grid(4).
__global__ __launch_bounds__(256) void k_gemm2(const _Float16* __restrict__ A,
                                               const _Float16* __restrict__ Bt,
                                               const float* __restrict__ br,
                                               const float* __restrict__ bc,
                                               float* __restrict__ Out)
{
    const int K = NDH;
    int wave = threadIdx.x >> 5, lane = threadIdx.x & 31;
    int laneHi = lane >> 4, l16 = lane & 15;
    int tile = blockIdx.x * 8 + wave;
    int mt = tile >> 1, nt = tile & 1;
    const _Float16* ap = A  + (size_t)(mt * 16 + l16) * K + laneHi * 16;
    const _Float16* bp = Bt + (size_t)(nt * 16 + l16) * K + laneHi * 16;
    v8f acc = {};
    for (int k0 = 0; k0 < K; k0 += 32) {
        v8h alo = *(const v8h*)(ap + k0);
        v8h ahi = *(const v8h*)(ap + k0 + 8);
        v16h bfr = *(const v16h*)(bp + k0);
        v16h a;
        #pragma unroll
        for (int t = 0; t < 8; ++t) { a[t] = alo[t]; a[t + 8] = ahi[t]; }
        acc = __builtin_amdgcn_wmma_f32_16x16x32_f16(false, a, false, bfr,
                                                     (short)0, acc, false, false);
    }
    #pragma unroll
    for (int i = 0; i < 8; ++i) {
        int m = mt * 16 + i + laneHi * 8;
        int n = nt * 16 + l16;
        float bias = (n < 4) ? br[n] : ((n < 4 + NCLS) ? bc[n - 4] : 0.f);
        Out[(size_t)m * NRC + n] = acc[i] + bias;
    }
}

// ---- final reductions: cls loss, reg loss, acc, offset ---------------------
__global__ void k_final(const float* __restrict__ rrc, const int* __restrict__ cls_s,
                        const float* __restrict__ tgt, float* __restrict__ out)
{
    __shared__ float sc[256], sr[256], sa[256], so[256];
    int i = threadIdx.x;                 // row 0..255
    const float* row = rrc + (size_t)i * NRC;
    float mx = row[4];
    for (int c = 1; c < NCLS; ++c) mx = fmaxf(mx, row[4 + c]);
    float se = 0.f;
    for (int c = 0; c < NCLS; ++c) se += expf(row[4 + c] - mx);
    int cls = cls_s[i];
    float logp = row[4 + cls] - mx - logf(se);
    float reg = 0.f, offv = 0.f, acc = 0.f;
    if (i < NPOS) {
        int am = 0; float av = row[4];
        for (int c = 1; c < NCLS; ++c)
            if (row[4 + c] > av) { av = row[4 + c]; am = c; }   // first-max like argmax
        acc = (am == cls) ? 1.f : 0.f;
        for (int c = 0; c < 4; ++c) {
            float d = fabsf(row[c] - tgt[i * 4 + c]);
            reg  += (d < 1.f) ? 0.5f * d * d : (d - 0.5f);
            offv += d;
        }
    }
    sc[i] = -logp; sr[i] = reg; sa[i] = acc; so[i] = offv;
    __syncthreads();
    for (int off = 128; off > 0; off >>= 1) {
        if (i < off) { sc[i]+=sc[i+off]; sr[i]+=sr[i+off]; sa[i]+=sa[i+off]; so[i]+=so[i+off]; }
        __syncthreads();
    }
    if (i == 0) {
        out[2] = sc[0] / 256.f;                  // rcnn_cls_loss
        out[3] = sr[0] / (float)(NPOS * 4);      // sum_c mean_i / 4
        out[4] = sa[0] / (float)NPOS;            // accuracy
        out[5] = so[0] / (float)(NPOS * 4);      // offset
    }
}

extern "C" void kernel_launch(void* const* d_in, const int* in_sizes, int n_in,
                              void* d_out, int out_size, void* d_ws, size_t ws_size,
                              hipStream_t stream)
{
    (void)in_sizes; (void)n_in; (void)out_size; (void)ws_size;
    const float* nms_reg      = (const float*)d_in[0];
    /* d_in[1] = nms_cls: unused by the reference forward pass */
    const float* feature_maps = (const float*)d_in[2];
    const float* bboxes       = (const float*)d_in[3];
    const int*   classes      = (const int*)  d_in[4];
    const float* anchors      = (const float*)d_in[5];
    const float* rpn_reg      = (const float*)d_in[6];
    const float* rpn_cls      = (const float*)d_in[7];
    const float* W1           = (const float*)d_in[8];
    const float* b1           = (const float*)d_in[9];
    const float* Wr           = (const float*)d_in[10];
    const float* br           = (const float*)d_in[11];
    const float* Wc           = (const float*)d_in[12];
    const float* bc           = (const float*)d_in[13];
    float* out = (float*)d_out;

    // Workspace carve-out (~34 MB), 256B-aligned slabs; everything rewritten per call.
    char* wsb = (char*)d_ws;
    size_t off = 0;
    auto alloc = [&](size_t bytes) -> void* {
        void* p = wsb + off;
        off += (bytes + 255) & ~(size_t)255;
        return p;
    };
    float*    max_iou = (float*)alloc((size_t)NB * NA * sizeof(float));
    int*      tidx    = (int*)  alloc((size_t)NB * NA * sizeof(int));
    float*    iou_m   = (float*)alloc((size_t)NB * NR * sizeof(float));
    int*      ridx    = (int*)  alloc((size_t)NB * NR * sizeof(int));
    float*    negsc   = (float*)alloc((size_t)NB * NR * sizeof(float));
    int*      pos     = (int*)  alloc(128  * sizeof(int));
    int*      neg     = (int*)  alloc(252  * sizeof(int));
    int*      posr    = (int*)  alloc(NPOS * sizeof(int));
    int*      negr    = (int*)  alloc(NNEG * sizeof(int));
    int*      cls_s   = (int*)  alloc(NROI * sizeof(int));
    float*    rois    = (float*)alloc(NROI * 5 * sizeof(float));
    float*    tgt     = (float*)alloc(NPOS * 4 * sizeof(float));
    _Float16* feat16  = (_Float16*)alloc((size_t)NROI * ND  * sizeof(_Float16));
    _Float16* W1t     = (_Float16*)alloc((size_t)ND   * NDH * sizeof(_Float16));
    _Float16* hmid16  = (_Float16*)alloc((size_t)NROI * NDH * sizeof(_Float16));
    _Float16* Wrct    = (_Float16*)alloc((size_t)NRC  * NDH * sizeof(_Float16));
    float*    rrc     = (float*)  alloc((size_t)NROI  * NRC * sizeof(float));

    // RPN branch
    k_rpn_iou<<<dim3((NB * NA) / 256), 256, 0, stream>>>(bboxes, anchors, max_iou, tidx);
    k_select_first<<<1, 1024, 0, stream>>>(max_iou, NB * NA, 0.5f, pos, 128, neg, 252);
    k_rpn_losses<<<1, 512, 0, stream>>>(pos, neg, rpn_cls, rpn_reg, bboxes, tidx, anchors, out);

    // R-CNN head sampling
    k_nms_iou<<<dim3((NB * NR + 255) / 256), 256, 0, stream>>>(nms_reg, bboxes, iou_m, ridx, negsc);
    k_select_first<<<1, 1024, 0, stream>>>(iou_m, NB * NR, 0.5f, posr, NPOS, negr, 0);
    k_topk<<<1, 1024, 0, stream>>>(negsc, NB * NR, negr, NNEG);
    k_build_rois<<<1, NROI, 0, stream>>>(posr, negr, nms_reg, classes, ridx, bboxes,
                                         rois, cls_s, tgt);

    // ROI pooling straight into the f16 GEMM A-panel
    k_roi_pool<<<dim3(NROI, NPP * NPP), NC, 0, stream>>>(feature_maps, rois, feat16);

    // Weight prep: f16, N-major panels (W1t fits in the 192MB L2 -> one HBM pass)
    k_w1_to_f16t<<<dim3(ND / 32, NDH / 32), 256, 0, stream>>>(W1, W1t);
    k_pack_wrc<<<NRC, NDH, 0, stream>>>(Wr, Wc, Wrct);

    // FC layers on v_wmma_f32_16x16x32_f16
    k_gemm1<<<dim3(NROI / 16, NDH / 128), 256, 0, stream>>>(feat16, W1t, b1, hmid16);
    k_gemm2<<<4, 256, 0, stream>>>(hmid16, Wrct, br, bc, rrc);

    // Losses / metrics
    k_final<<<1, NROI, 0, stream>>>(rrc, cls_s, tgt, out);
}
